// MultiHeadAttention_67860483277471
// MI455X (gfx1250) — matrix operand order repaired and checked by reading, not compiled, not run on hardware
//
#include <hip/hip_runtime.h>
#include <hip/hip_bf16.h>

// ---------------- problem constants ----------------
#define BATCH 4
#define SEQ   1024
#define DIMM  512
#define NHEAD 8
#define KDIM  256          // per-head q/k dim
#define DHEAD 64           // per-head v dim
#define NQK   2048         // KDIM*NHEAD
#define MROWS 4096         // BATCH*SEQ
#define EPSLN 1e-5f
#define ATT_SCALE 4.0f     // 256^0.25

typedef __attribute__((ext_vector_type(16))) __bf16       v16bf;
typedef __attribute__((ext_vector_type(8)))  float        v8f;
typedef __attribute__((ext_vector_type(8)))  unsigned int v8u;
typedef __attribute__((ext_vector_type(4)))  int          v4i;

// ---------------- gfx1250 async global->LDS copy ----------------
#if defined(__has_builtin)
#  if __has_builtin(__builtin_amdgcn_global_load_async_to_lds_b128)
#    define HAVE_ASYNC 1
#  endif
#endif
#ifndef HAVE_ASYNC
#  define HAVE_ASYNC 0
#endif

__device__ __forceinline__ void cp_async_b128(const void* g, void* l) {
#if HAVE_ASYNC
    __builtin_amdgcn_global_load_async_to_lds_b128(
        (__attribute__((address_space(1))) v4i*)g,
        (__attribute__((address_space(3))) v4i*)l,
        /*offset=*/0, /*cpol=*/0);
#else
    *(uint4*)l = *(const uint4*)g;
#endif
}

__device__ __forceinline__ void cp_async_wait() {
#if HAVE_ASYNC
#  if defined(__has_builtin) && __has_builtin(__builtin_amdgcn_s_wait_asynccnt)
    __builtin_amdgcn_s_wait_asynccnt(0);
#  else
    asm volatile("s_wait_asynccnt 0x0" ::: "memory");
#  endif
#endif
}

__device__ __forceinline__ unsigned short f2bf(float f) {
    unsigned int u = __builtin_bit_cast(unsigned int, f);
    unsigned int r = u + 0x7FFFu + ((u >> 16) & 1u);
    return (unsigned short)(r >> 16);
}

__device__ __forceinline__ v8f wmma_bf16(v16bf a, v16bf b, v8f c) {
    return __builtin_amdgcn_wmma_f32_16x16x32_bf16(
        /*neg_a=*/false, a, /*neg_b=*/false, b,
        /*c_mod=*/(short)0, c, /*reuse_a=*/false, /*reuse_b=*/false);
}

// A-fragment (16 rows x 32 K, bf16). p -> element [0][0], element [m][k] at p[m*stride+k].
__device__ __forceinline__ v16bf frag_a_ld(const unsigned short* p, int stride) {
    int lane = threadIdx.x & 31;
    int m = lane & 15, h = lane >> 4;
    const unsigned short* r = p + m * stride;
    v8u u;
#pragma unroll
    for (int j = 0; j < 8; ++j) {
        int kk = ((j >> 2) << 4) + (h << 3) + ((j & 3) << 1);
        u[j] = *(const unsigned int*)(r + kk);
    }
    return __builtin_bit_cast(v16bf, u);
}

// B-fragment (32 K x 16 cols, bf16). Element [k][n] stored at p[n*stride + k]
// (source is K-contiguous / pre-transposed).
__device__ __forceinline__ v16bf frag_b_ld(const unsigned short* p, int stride) {
    int lane = threadIdx.x & 31;
    int n = lane & 15, h = lane >> 4;
    const unsigned short* r = p + n * stride;
    v8u u;
#pragma unroll
    for (int j = 0; j < 8; ++j) {
        int kk = ((j >> 2) << 4) + (h << 3) + ((j & 3) << 1);
        u[j] = *(const unsigned int*)(r + kk);
    }
    return __builtin_bit_cast(v16bf, u);
}

// ---------------- conversion kernels ----------------
__global__ void k_cvt_bf16(const float* __restrict__ x, unsigned short* __restrict__ y, int n) {
    for (int i = blockIdx.x * blockDim.x + threadIdx.x; i < n; i += gridDim.x * blockDim.x)
        y[i] = f2bf(x[i]);
}

// wt[n*K + k] = bf16(w[k*N + n])  (transpose + convert, W is K x N row-major)
__global__ void k_cvt_tr(const float* __restrict__ w, unsigned short* __restrict__ wt, int K, int N) {
    int i = blockIdx.x * blockDim.x + threadIdx.x;
    if (i >= K * N) return;
    int n = i / K, k = i - n * K;
    wt[i] = f2bf(w[k * N + n]);
}

// ---------------- GEMM: C(f32,MxN) = A(bf16,MxK) @ Wt(bf16,NxK)^T + bias ----------------
// macro tile 128(M) x 128(N), staged K-step 64, double-buffered async LDS pipeline,
// 8 waves each computing 32x64 (16 WMMAs per staged step).
__global__ __launch_bounds__(256) void k_gemm(const unsigned short* __restrict__ A,
                                              const unsigned short* __restrict__ Wt,
                                              const float* __restrict__ bias,
                                              float* __restrict__ C,
                                              int M, int N, int K) {
    __shared__ alignas(16) unsigned short lA[2][128 * 64];   // 2 x 16 KB
    __shared__ alignas(16) unsigned short lB[2][128 * 64];   // 2 x 16 KB
    const int nb = blockIdx.x * 128, mb = blockIdx.y * 128;
    const int t = threadIdx.x, w = t >> 5, lane = t & 31;
    const int wm = (w >> 1) * 32, wn = (w & 1) * 64;

    v8f acc[2][4] = {};

    auto stage = [&](int kb, int buf) {
#pragma unroll
        for (int p = 0; p < 4; ++p) {
            int f = p * 2048 + t * 8;       // 8 bf16 = 16 B per lane per issue
            int r = f >> 6, c = f & 63;
            cp_async_b128(A  + (mb + r) * K + kb + c, &lA[buf][f]);
            cp_async_b128(Wt + (nb + r) * K + kb + c, &lB[buf][f]);
        }
    };

    stage(0, 0);
    cp_async_wait();
    __syncthreads();

    int buf = 0;
    for (int kb = 0; kb < K; kb += 64) {
        if (kb + 64 < K) stage(kb + 64, buf ^ 1);   // prefetch next tile into other buffer
#pragma unroll
        for (int ks = 0; ks < 2; ++ks) {
            v16bf af[2], bfg[4];
#pragma unroll
            for (int mt = 0; mt < 2; ++mt)
                af[mt]  = frag_a_ld(&lA[buf][(wm + mt * 16) * 64 + ks * 32], 64);
#pragma unroll
            for (int nt = 0; nt < 4; ++nt)
                bfg[nt] = frag_b_ld(&lB[buf][(wn + nt * 16) * 64 + ks * 32], 64);
#pragma unroll
            for (int mt = 0; mt < 2; ++mt)
#pragma unroll
                for (int nt = 0; nt < 4; ++nt)
                    acc[mt][nt] = wmma_bf16(af[mt], bfg[nt], acc[mt][nt]);
        }
        cp_async_wait();
        __syncthreads();
        buf ^= 1;
    }

    const int n0 = lane & 15, half = lane >> 4;
#pragma unroll
    for (int mt = 0; mt < 2; ++mt)
#pragma unroll
        for (int nt = 0; nt < 4; ++nt) {
            int col = nb + wn + nt * 16 + n0;
            float bv = bias[col];
            int base = (mb + wm + mt * 16 + half * 8) * N + col;
#pragma unroll
            for (int r = 0; r < 8; ++r)
                C[base + r * N] = acc[mt][nt][r] + bv;
        }
}

// ---------------- row LayerNorm: out = LN(X_row)*g + beta (+ addin), bf16/f32 out ----------------
__global__ __launch_bounds__(256) void k_layernorm(const float* __restrict__ X,
                                                   const float* __restrict__ g,
                                                   const float* __restrict__ beta,
                                                   const float* __restrict__ addin,
                                                   unsigned short* __restrict__ outb,
                                                   float* __restrict__ outf,
                                                   int N) {
    __shared__ float red[256];
    const int row = blockIdx.x;
    const float* x = X + row * N;

    float s = 0.f, ss = 0.f;
    for (int i = threadIdx.x; i < N; i += 256) { float v = x[i]; s += v; ss += v * v; }

    red[threadIdx.x] = s;  __syncthreads();
    for (int o = 128; o > 0; o >>= 1) { if (threadIdx.x < o) red[threadIdx.x] += red[threadIdx.x + o]; __syncthreads(); }
    float mean = red[0] / (float)N;  __syncthreads();
    red[threadIdx.x] = ss; __syncthreads();
    for (int o = 128; o > 0; o >>= 1) { if (threadIdx.x < o) red[threadIdx.x] += red[threadIdx.x + o]; __syncthreads(); }
    float var  = red[0] / (float)N - mean * mean;
    float rstd = rsqrtf(var + EPSLN);

    for (int i = threadIdx.x; i < N; i += 256) {
        float y = (x[i] - mean) * rstd * g[i] + beta[i];
        if (addin) y += addin[row * N + i];
        if (outb)  outb[row * N + i] = f2bf(y);
        if (outf)  outf[row * N + i] = y;
    }
}

// ---------------- permutes ----------------
// pe_in[b][r][c] = Vn[b][ (r&127)*8 + (c>>6) ][ (r>>7)*64 + (c&63) ]   (head-major reshape of V)
__global__ void k_permute_pe(const unsigned short* __restrict__ Vn, unsigned short* __restrict__ peIn) {
    int i = blockIdx.x * blockDim.x + threadIdx.x;   // BATCH*1024*512 total
    int c = i & 511, r = (i >> 9) & 1023, b = i >> 19;
    int h = r >> 7, s0 = r & 127, s8 = c >> 6, dh = c & 63;
    peIn[i] = Vn[(b * SEQ + s0 * 8 + s8) * DIMM + h * DHEAD + dh];
}

// Vt[b][h][dh][s] = Vn[b*SEQ+s][h*64+dh]
__global__ void k_permute_vt(const unsigned short* __restrict__ Vn, unsigned short* __restrict__ Vt) {
    int i = blockIdx.x * blockDim.x + threadIdx.x;   // BATCH*NH*64*1024 total
    int s  = i & 1023;
    int dh = (i >> 10) & 63;
    int h  = (i >> 16) & 7;
    int b  = i >> 19;
    Vt[i] = Vn[(b * SEQ + s) * DIMM + h * DHEAD + dh];
}

// ---------------- flash attention ----------------
// grid: (SEQ/64, BATCH*NHEAD), block: 128 (4 waves, 16 queries/wave)
__global__ __launch_bounds__(128) void k_attention(const unsigned short* __restrict__ Qn,
                                                   const unsigned short* __restrict__ Kn,
                                                   const unsigned short* __restrict__ Vt,
                                                   float* __restrict__ ctx) {
    __shared__ alignas(16) unsigned short lQ[64 * 256];      // 32 KB
    __shared__ alignas(16) unsigned short lK[64 * 256];      // 32 KB
    __shared__ alignas(16) unsigned short lV[64 * 64];       //  8 KB  Vt block [dh][key]
    __shared__ alignas(16) unsigned short lP[4][16 * 64];    //  8 KB  per-wave P staging

    const int bh = blockIdx.y;
    const int b = bh / NHEAD, h = bh % NHEAD;
    const int q0 = blockIdx.x * 64;
    const int t = threadIdx.x, w = t >> 5, lane = t & 31;
    const int n0 = lane & 15, half = lane >> 4;

    // async-load Q block (64 x 256 bf16); completion covered by first in-loop wait
#pragma unroll
    for (int p = 0; p < 16; ++p) {
        int f = p * 1024 + t * 8;
        int r = f >> 8, c = f & 255;
        cp_async_b128(Qn + (b * SEQ + q0 + r) * NQK + h * KDIM + c, lQ + f);
    }

    float m_r[8], l_r[8];
    v8f o[4] = {};
#pragma unroll
    for (int r = 0; r < 8; ++r) { m_r[r] = -3.0e38f; l_r[r] = 0.f; }

    for (int j0 = 0; j0 < SEQ; j0 += 64) {
        __syncthreads();   // protect lK/lV against overwrite while still in use
        // async-load K block 64x256 and Vt block 64x64
#pragma unroll
        for (int p = 0; p < 16; ++p) {
            int f = p * 1024 + t * 8;
            int r = f >> 8, c = f & 255;
            cp_async_b128(Kn + (b * SEQ + j0 + r) * NQK + h * KDIM + c, lK + f);
        }
#pragma unroll
        for (int p = 0; p < 4; ++p) {
            int f = p * 1024 + t * 8;
            int r = f >> 6, c = f & 63;
            cp_async_b128(Vt + ((b * NHEAD + h) * DHEAD + r) * SEQ + j0 + c, lV + f);
        }
        cp_async_wait();
        __syncthreads();

        // scores: 16 queries x 64 keys, K-dim 256 = 8 wmma steps
        v8f sc[4] = {};
#pragma unroll
        for (int ks = 0; ks < 8; ++ks) {
            v16bf a = frag_a_ld(lQ + (w * 16) * 256 + ks * 32, 256);
#pragma unroll
            for (int nt = 0; nt < 4; ++nt) {
                v16bf bb = frag_b_ld(lK + (nt * 16) * 256 + ks * 32, 256);
                sc[nt] = wmma_bf16(a, bb, sc[nt]);
            }
        }

        // online softmax (rows r+half*8, row values across 16 n-lanes x 4 tiles)
        unsigned short* pw = lP[w];
#pragma unroll
        for (int r = 0; r < 8; ++r) {
            float mx = sc[0][r] * ATT_SCALE;
#pragma unroll
            for (int nt = 0; nt < 4; ++nt) {
                sc[nt][r] *= ATT_SCALE;
                mx = fmaxf(mx, sc[nt][r]);
            }
#pragma unroll
            for (int off = 1; off <= 8; off <<= 1)
                mx = fmaxf(mx, __shfl_xor(mx, off, 32));
            float mnew = fmaxf(m_r[r], mx);
            float corr = __expf(m_r[r] - mnew);
            m_r[r] = mnew;
            l_r[r] *= corr;
#pragma unroll
            for (int dt = 0; dt < 4; ++dt) o[dt][r] *= corr;
            float psum = 0.f;
#pragma unroll
            for (int nt = 0; nt < 4; ++nt) {
                float p = __expf(sc[nt][r] - mnew);
                sc[nt][r] = p;
                psum += p;
                pw[(r + half * 8) * 64 + nt * 16 + n0] = f2bf(p);
            }
#pragma unroll
            for (int off = 1; off <= 8; off <<= 1)
                psum += __shfl_xor(psum, off, 32);
            l_r[r] += psum;
        }

        // O += P(16x64) @ V(64x64); keys dim = 2 wmma steps
#pragma unroll
        for (int ks = 0; ks < 2; ++ks) {
            v16bf a = frag_a_ld(pw + ks * 32, 64);
#pragma unroll
            for (int dt = 0; dt < 4; ++dt) {
                v16bf bb = frag_b_ld(lV + (dt * 16) * 64 + ks * 32, 64);
                o[dt] = wmma_bf16(a, bb, o[dt]);
            }
        }
    }

    // epilogue: ctx[b][q][h*64+dh] = O / l
#pragma unroll
    for (int r = 0; r < 8; ++r) {
        float inv = 1.0f / l_r[r];
        int q = q0 + w * 16 + r + half * 8;
#pragma unroll
        for (int dt = 0; dt < 4; ++dt) {
            ctx[(b * SEQ + q) * DIMM + h * DHEAD + dt * 16 + n0] = o[dt][r] * inv;
        }
    }
}

// ---------------- launcher ----------------
extern "C" void kernel_launch(void* const* d_in, const int* in_sizes, int n_in,
                              void* d_out, int out_size, void* d_ws, size_t ws_size,
                              hipStream_t stream) {
    (void)in_sizes; (void)n_in; (void)out_size; (void)ws_size;
    const float* q      = (const float*)d_in[0];
    const float* k      = (const float*)d_in[1];
    const float* v      = (const float*)d_in[2];
    const float* wq_w   = (const float*)d_in[3];
    const float* wq_b   = (const float*)d_in[4];
    const float* wq_g   = (const float*)d_in[5];
    const float* wq_be  = (const float*)d_in[6];
    const float* wk_w   = (const float*)d_in[7];
    const float* wk_b   = (const float*)d_in[8];
    const float* wk_g   = (const float*)d_in[9];
    const float* wk_be  = (const float*)d_in[10];
    const float* wv_w   = (const float*)d_in[11];
    const float* wv_b   = (const float*)d_in[12];
    const float* wv_g   = (const float*)d_in[13];
    const float* wv_be  = (const float*)d_in[14];
    const float* pe_w   = (const float*)d_in[15];
    const float* pe_b   = (const float*)d_in[16];
    const float* pe_g   = (const float*)d_in[17];
    const float* pe_be  = (const float*)d_in[18];
    const float* out_w  = (const float*)d_in[19];
    const float* out_b  = (const float*)d_in[20];
    const float* out_g  = (const float*)d_in[21];
    const float* out_be = (const float*)d_in[22];
    float* dout = (float*)d_out;

    char* ws = (char*)d_ws;
    auto give = [&](size_t bytes) -> void* {
        void* p = (void*)ws;
        ws += (bytes + 255) & ~(size_t)255;
        return p;
    };
    unsigned short* qb    = (unsigned short*)give((size_t)MROWS * DIMM * 2);
    unsigned short* kb    = (unsigned short*)give((size_t)MROWS * DIMM * 2);
    unsigned short* vb    = (unsigned short*)give((size_t)MROWS * DIMM * 2);
    unsigned short* WtQ   = (unsigned short*)give((size_t)NQK  * DIMM * 2);
    unsigned short* WtK   = (unsigned short*)give((size_t)NQK  * DIMM * 2);
    unsigned short* WtV   = (unsigned short*)give((size_t)DIMM * DIMM * 2);
    unsigned short* WtPE  = (unsigned short*)give((size_t)DIMM * DIMM * 2);
    unsigned short* WtOUT = (unsigned short*)give((size_t)DIMM * DIMM * 2);
    float*          pre   = (float*)         give((size_t)MROWS * NQK * 4);   // reused
    unsigned short* Qn    = (unsigned short*)give((size_t)MROWS * NQK * 2);
    unsigned short* Kn    = (unsigned short*)give((size_t)MROWS * NQK * 2);
    unsigned short* Vn    = (unsigned short*)give((size_t)MROWS * DIMM * 2);
    unsigned short* peIn  = (unsigned short*)give((size_t)MROWS * DIMM * 2);
    unsigned short* Vt    = (unsigned short*)give((size_t)MROWS * DIMM * 2);
    float*          ctx   = (float*)         give((size_t)MROWS * DIMM * 4);
    unsigned short* ctx2  = (unsigned short*)give((size_t)MROWS * DIMM * 2);

    const int NELT = MROWS * DIMM;   // 2,097,152

    // 1) bf16 conversions
    k_cvt_bf16<<<1024, 256, 0, stream>>>(q, qb, NELT);
    k_cvt_bf16<<<1024, 256, 0, stream>>>(k, kb, NELT);
    k_cvt_bf16<<<1024, 256, 0, stream>>>(v, vb, NELT);
    k_cvt_tr<<<(DIMM * NQK) / 256, 256, 0, stream>>>(wq_w, WtQ, DIMM, NQK);
    k_cvt_tr<<<(DIMM * NQK) / 256, 256, 0, stream>>>(wk_w, WtK, DIMM, NQK);
    k_cvt_tr<<<(DIMM * DIMM) / 256, 256, 0, stream>>>(wv_w, WtV, DIMM, DIMM);
    k_cvt_tr<<<(DIMM * DIMM) / 256, 256, 0, stream>>>(pe_w, WtPE, DIMM, DIMM);
    k_cvt_tr<<<(DIMM * DIMM) / 256, 256, 0, stream>>>(out_w, WtOUT, DIMM, DIMM);

    // 2) Q/K/V projections + LN
    k_gemm<<<dim3(NQK / 128, MROWS / 128), 256, 0, stream>>>(qb, WtQ, wq_b, pre, MROWS, NQK, DIMM);
    k_layernorm<<<MROWS, 256, 0, stream>>>(pre, wq_g, wq_be, nullptr, Qn, nullptr, NQK);
    k_gemm<<<dim3(NQK / 128, MROWS / 128), 256, 0, stream>>>(kb, WtK, wk_b, pre, MROWS, NQK, DIMM);
    k_layernorm<<<MROWS, 256, 0, stream>>>(pre, wk_g, wk_be, nullptr, Kn, nullptr, NQK);
    k_gemm<<<dim3(DIMM / 128, MROWS / 128), 256, 0, stream>>>(vb, WtV, wv_b, pre, MROWS, DIMM, DIMM);
    k_layernorm<<<MROWS, 256, 0, stream>>>(pre, wv_g, wv_be, nullptr, Vn, nullptr, DIMM);

    // 3) layouts for attention / pe path
    k_permute_pe<<<NELT / 256, 256, 0, stream>>>(Vn, peIn);
    k_permute_vt<<<NELT / 256, 256, 0, stream>>>(Vn, Vt);

    // 4) attention
    k_attention<<<dim3(SEQ / 64, BATCH * NHEAD), 128, 0, stream>>>(Qn, Kn, Vt, ctx);

    // 5) pe path: ctx2 = bf16( ctx + LN(peIn @ Wpe + b) )
    k_gemm<<<dim3(DIMM / 128, MROWS / 128), 256, 0, stream>>>(peIn, WtPE, pe_b, pre, MROWS, DIMM, DIMM);
    k_layernorm<<<MROWS, 256, 0, stream>>>(pre, pe_g, pe_be, ctx, ctx2, nullptr, DIMM);

    // 6) output projection + final LN -> d_out (fp32)
    k_gemm<<<dim3(DIMM / 128, MROWS / 128), 256, 0, stream>>>(ctx2, WtOUT, out_b, pre, MROWS, DIMM, DIMM);
    k_layernorm<<<MROWS, 256, 0, stream>>>(pre, out_g, out_be, nullptr, nullptr, dout, DIMM);
}
